// TransformerBlock_6270652252782
// MI455X (gfx1250) — matrix operand order repaired and checked by reading, not compiled, not run on hardware
//
#include <hip/hip_runtime.h>
#include <hip/hip_bf16.h>

// ---------------- types ----------------
typedef _Float16 v16h __attribute__((ext_vector_type(16)));
typedef _Float16 v8h  __attribute__((ext_vector_type(8)));
typedef float    v8f  __attribute__((ext_vector_type(8)));

static constexpr int Dm = 512;     // d_model
static constexpr int Tt = 4096;    // seq len
static constexpr int Bb = 2;       // batch
static constexpr int BT = Bb * Tt; // 8192 rows
static constexpr int Dff = 2048;   // 4*d

// ---------------- WMMA fragment loaders ----------------
// A operand: 16x32 f16, M = lane%16, K elems {koff..koff+7, 16+koff..16+koff+7}, koff = (lane<16)?0:8
__device__ __forceinline__ v16h loadA16(const _Float16* __restrict__ base, int ld) {
  const int lane = threadIdx.x & 31;
  const int row  = lane & 15;
  const int koff = (lane < 16) ? 0 : 8;
  const _Float16* p = base + (size_t)row * ld + koff;
  union { v16h v; v8h h[2]; } u;
  u.h[0] = *(const v8h*)(p);
  u.h[1] = *(const v8h*)(p + 16);
  return u.v;
}

// B operand: 32x16 f16, N = lane%16 (a row of the [N,K] array), K elems {koff..koff+7, 16+koff..16+koff+7}, koff = (lane<16)?0:8
__device__ __forceinline__ v16h loadB16(const _Float16* __restrict__ base, int ld) {
  const int lane = threadIdx.x & 31;
  const int col  = lane & 15;
  const int koff = (lane < 16) ? 0 : 8;
  const _Float16* p = base + (size_t)col * ld + koff;
  union { v16h v; v8h h[2]; } u;
  u.h[0] = *(const v8h*)(p);
  u.h[1] = *(const v8h*)(p + 16);
  return u.v;
}

__device__ __forceinline__ v8f wmma_f16(v16h a, v16h b, v8f c) {
  return __builtin_amdgcn_wmma_f32_16x16x32_f16(false, a, false, b, (short)0, c, false, false);
}

// ---------------- elementwise converts ----------------
__global__ void cvt_f32_to_f16(const float* __restrict__ in, _Float16* __restrict__ out, int n) {
  int i = blockIdx.x * blockDim.x + threadIdx.x;
  if (i < n) out[i] = (_Float16)in[i];
}

// in: [K,N] f32 row-major  ->  out: [N,K] f16 row-major
__global__ void transpose_f16(const float* __restrict__ in, _Float16* __restrict__ out, int K, int N) {
  int i = blockIdx.x * blockDim.x + threadIdx.x;
  if (i < K * N) {
    int k = i % K;
    int n = i / K;
    out[i] = (_Float16)in[(size_t)k * N + n];
  }
}

// ---------------- generic WMMA GEMM: out = A[M,K] @ Bt[N,K]^T + bias ----------------
// EPI: 0 = store f16 row-major, 1 = store f16 transposed [b][N][T], 2 = relu + f16 row-major, 3 = store f32 row-major
template <int EPI>
__global__ __launch_bounds__(256) void gemm_wmma(const _Float16* __restrict__ A,
                                                 const _Float16* __restrict__ Bt,
                                                 const float* __restrict__ bias,
                                                 void* __restrict__ out,
                                                 int M, int N, int K, int Tlen) {
  const int lane = threadIdx.x & 31;
  const int wave = threadIdx.x >> 5;
  const int mt = blockIdx.x;
  const int nc = blockIdx.y * 8 + wave;
  const int n0 = nc * 16;
  if (n0 >= N) return;
  const int cN = lane & 15;
  const float binit = bias[n0 + cN];
  v8f c;
#pragma unroll
  for (int r = 0; r < 8; ++r) c[r] = binit;

  const _Float16* Ab = A + (size_t)mt * 16 * K;
  const _Float16* Bb_ = Bt + (size_t)n0 * K;

  // manual 2-deep software pipeline: next fragments in flight during current WMMA
  v16h a0 = loadA16(Ab, K);
  v16h b0 = loadB16(Bb_, K);
  for (int k0 = 32; k0 < K; k0 += 32) {
    v16h a1 = loadA16(Ab + k0, K);
    v16h b1 = loadB16(Bb_ + k0, K);
    c = wmma_f16(a0, b0, c);
    a0 = a1;
    b0 = b1;
  }
  c = wmma_f16(a0, b0, c);

#pragma unroll
  for (int r = 0; r < 8; ++r) {
    const int Mr = r + ((lane < 16) ? 0 : 8);
    const int row = mt * 16 + Mr;
    float val = c[r];
    if (EPI == 2) val = fmaxf(val, 0.0f);
    if (EPI == 0 || EPI == 2) {
      ((_Float16*)out)[(size_t)row * N + n0 + cN] = (_Float16)val;
    } else if (EPI == 1) {
      const int b_ = row / Tlen;
      const int t  = row - b_ * Tlen;
      ((_Float16*)out)[((size_t)b_ * N + (n0 + cN)) * (size_t)Tlen + t] = (_Float16)val;
    } else { // EPI == 3
      ((float*)out)[(size_t)row * N + n0 + cN] = val;
    }
  }
}

// ---------------- flash attention (one wave per 16-query tile) ----------------
// launch_bounds(128, 2): cap at 512 VGPRs/wave so 2 waves/SIMD co-reside.
__global__ __launch_bounds__(128, 2) void attn_flash(const _Float16* __restrict__ qh,
                                                     const _Float16* __restrict__ kh,
                                                     const _Float16* __restrict__ vt,
                                                     const float* __restrict__ dsp,
                                                     float* __restrict__ attn_out) {
  __shared__ __align__(16) _Float16 sp[4 * 16 * 32]; // per-wave D->A transpose scratch
  const int lane = threadIdx.x & 31;
  const int wave = threadIdx.x >> 5;
  const int tile = wave * gridDim.x + blockIdx.x;   // 0..511, qi spread across blocks
  const int b  = tile / (Tt / 16);
  const int qi = tile % (Tt / 16);
  const float ds  = dsp[0];
  const float rsD = 0.044194173824159216f; // 1/sqrt(512)
  const int cN = lane & 15;
  const int Mhi = (lane < 16) ? 0 : 8;

  const _Float16* kbat = kh + (size_t)b * Tt * Dm;
  const _Float16* vbat = vt + (size_t)b * Dm * Tt;

  // cache the 16x512 q tile as A fragments in registers (16 * v16h)
  v16h qf[16];
  const _Float16* qbase = qh + (size_t)(b * Tt + qi * 16) * Dm;
#pragma unroll
  for (int kc = 0; kc < 16; ++kc) qf[kc] = loadA16(qbase + kc * 32, Dm);

  v8f O[32];
#pragma unroll
  for (int n = 0; n < 32; ++n)
#pragma unroll
    for (int r = 0; r < 8; ++r) O[n][r] = 0.0f;

  float mrun[8], lrun[8];
#pragma unroll
  for (int r = 0; r < 8; ++r) { mrun[r] = -INFINITY; lrun[r] = 0.0f; }

  _Float16* spw = sp + wave * 512;
  const int kpCount = qi / 2 + 1; // covers keys [0, 16*qi+16)

  for (int kp = 0; kp < kpCount; ++kp) {
    // prefetch next key tile (one row per lane, first cacheline)
    if (kp + 1 < kpCount) {
      const _Float16* nk = kbat + (size_t)((kp + 1) * 32 + lane) * Dm;
      __builtin_prefetch(nk, 0, 3);
    }

    // ---- scores for two 16-key tiles: two independent WMMA chains,
    //      manually double-buffered B fragments ----
    const _Float16* kbase0 = kbat + (size_t)(kp * 32) * Dm;
    const _Float16* kbase1 = kbat + (size_t)(kp * 32 + 16) * Dm;
    v8f c0, c1;
#pragma unroll
    for (int r = 0; r < 8; ++r) { c0[r] = 0.0f; c1[r] = 0.0f; }
    v16h bk0 = loadB16(kbase0, Dm);
    v16h bk1 = loadB16(kbase1, Dm);
#pragma unroll
    for (int kc = 0; kc < 16; ++kc) {
      v16h nb0, nb1;
      if (kc < 15) {
        nb0 = loadB16(kbase0 + (kc + 1) * 32, Dm);
        nb1 = loadB16(kbase1 + (kc + 1) * 32, Dm);
      }
      c0 = wmma_f16(qf[kc], bk0, c0);
      c1 = wmma_f16(qf[kc], bk1, c1);
      if (kc < 15) { bk0 = nb0; bk1 = nb1; }
    }

    float s[2][8];
#pragma unroll
    for (int r = 0; r < 8; ++r) {
      const int qidx = qi * 16 + r + Mhi;
      const int k0 = kp * 32 + cN;
      const int k1 = k0 + 16;
      float s0 = (c0[r] - ds * fabsf((float)(qidx - k0))) * rsD;
      float s1 = (c1[r] - ds * fabsf((float)(qidx - k1))) * rsD;
      if (k0 > qidx) s0 = -INFINITY;
      if (k1 > qidx) s1 = -INFINITY;
      s[0][r] = s0;
      s[1][r] = s1;
    }

    // ---- online softmax update ----
    float alpha[8];
#pragma unroll
    for (int r = 0; r < 8; ++r) {
      float tmax = fmaxf(s[0][r], s[1][r]);
      tmax = fmaxf(tmax, __shfl_xor(tmax, 1));
      tmax = fmaxf(tmax, __shfl_xor(tmax, 2));
      tmax = fmaxf(tmax, __shfl_xor(tmax, 4));
      tmax = fmaxf(tmax, __shfl_xor(tmax, 8));
      const float mnew = fmaxf(mrun[r], tmax);
      alpha[r] = __expf(mrun[r] - mnew);
      const float p0 = __expf(s[0][r] - mnew);
      const float p1 = __expf(s[1][r] - mnew);
      s[0][r] = p0; s[1][r] = p1;
      float rs = p0 + p1;
      rs += __shfl_xor(rs, 1);
      rs += __shfl_xor(rs, 2);
      rs += __shfl_xor(rs, 4);
      rs += __shfl_xor(rs, 8);
      lrun[r] = lrun[r] * alpha[r] + rs;
      mrun[r] = mnew;
    }

    // ---- D-layout probs -> LDS -> A-layout fragment ----
#pragma unroll
    for (int h = 0; h < 2; ++h)
#pragma unroll
      for (int r = 0; r < 8; ++r)
        spw[(r + Mhi) * 32 + h * 16 + cN] = (_Float16)s[h][r];
    asm volatile("s_wait_dscnt 0" ::: "memory");
    v16h pA = loadA16(spw, 32);

    // ---- O = alpha*O + P @ V (double-buffered V fragments) ----
    const _Float16* vbase = vbat + (size_t)kp * 32;
    v16h vb = loadB16(vbase, Tt);
#pragma unroll
    for (int n = 0; n < 32; ++n) {
      v16h vn;
      if (n < 31) vn = loadB16(vbase + (size_t)(n + 1) * 16 * Tt, Tt);
      v8f cc = O[n];
#pragma unroll
      for (int r = 0; r < 8; ++r) cc[r] *= alpha[r];
      O[n] = wmma_f16(pA, vb, cc);
      if (n < 31) vb = vn;
    }
  }

  float inv[8];
#pragma unroll
  for (int r = 0; r < 8; ++r) inv[r] = 1.0f / lrun[r];
#pragma unroll
  for (int n = 0; n < 32; ++n)
#pragma unroll
    for (int r = 0; r < 8; ++r) {
      const int row = b * Tt + qi * 16 + r + Mhi;
      attn_out[(size_t)row * Dm + n * 16 + cN] = O[n][r] * inv[r];
    }
}

// ---------------- layernorm over rows of d=512: out = LN(a+b)*g+be ----------------
__global__ __launch_bounds__(256) void ln_rows(const float* __restrict__ a,
                                               const float* __restrict__ bsrc,
                                               const float* __restrict__ g,
                                               const float* __restrict__ be,
                                               float* __restrict__ out,
                                               _Float16* __restrict__ outh,
                                               int rows) {
  const int lane = threadIdx.x & 31;
  const int wave = threadIdx.x >> 5;
  const int row = blockIdx.x * 8 + wave;
  if (row >= rows) return;
  const float* pa = a + (size_t)row * Dm;
  const float* pb = bsrc + (size_t)row * Dm;
  float u[16];
  float s = 0.0f;
#pragma unroll
  for (int j = 0; j < 16; ++j) {
    const int i = j * 32 + lane;
    u[j] = pa[i] + pb[i];
    s += u[j];
  }
  s += __shfl_xor(s, 1); s += __shfl_xor(s, 2); s += __shfl_xor(s, 4);
  s += __shfl_xor(s, 8); s += __shfl_xor(s, 16);
  const float mu = s * (1.0f / Dm);
  float v = 0.0f;
#pragma unroll
  for (int j = 0; j < 16; ++j) { const float d = u[j] - mu; v += d * d; }
  v += __shfl_xor(v, 1); v += __shfl_xor(v, 2); v += __shfl_xor(v, 4);
  v += __shfl_xor(v, 8); v += __shfl_xor(v, 16);
  const float rs = rsqrtf(v * (1.0f / Dm) + 1e-5f);
#pragma unroll
  for (int j = 0; j < 16; ++j) {
    const int i = j * 32 + lane;
    const float z = (u[j] - mu) * rs * g[i] + be[i];
    out[(size_t)row * Dm + i] = z;
    if (outh) outh[(size_t)row * Dm + i] = (_Float16)z;
  }
}

// ---------------- host ----------------
extern "C" void kernel_launch(void* const* d_in, const int* in_sizes, int n_in,
                              void* d_out, int out_size, void* d_ws, size_t ws_size,
                              hipStream_t stream) {
  const float* tokens = (const float*)d_in[0];
  const float* Wq = (const float*)d_in[1];
  const float* bq = (const float*)d_in[2];
  const float* Wk = (const float*)d_in[3];
  const float* bk = (const float*)d_in[4];
  const float* Wv = (const float*)d_in[5];
  const float* bv = (const float*)d_in[6];
  const float* dscale = (const float*)d_in[7];
  const float* W1 = (const float*)d_in[8];
  const float* b1 = (const float*)d_in[9];
  const float* W2 = (const float*)d_in[10];
  const float* b2 = (const float*)d_in[11];
  const float* g1 = (const float*)d_in[12];
  const float* be1 = (const float*)d_in[13];
  const float* g2 = (const float*)d_in[14];
  const float* be2 = (const float*)d_in[15];

  // workspace bump allocator (256B aligned)
  size_t off = 0;
  auto alloc = [&](size_t bytes) -> char* {
    size_t o = (off + 255) & ~(size_t)255;
    off = o + bytes;
    return (char*)d_ws + o;
  };
  _Float16* tokh = (_Float16*)alloc((size_t)BT * Dm * 2);
  _Float16* WqT  = (_Float16*)alloc((size_t)Dm * Dm * 2);
  _Float16* WkT  = (_Float16*)alloc((size_t)Dm * Dm * 2);
  _Float16* WvT  = (_Float16*)alloc((size_t)Dm * Dm * 2);
  _Float16* W1T  = (_Float16*)alloc((size_t)Dm * Dff * 2);
  _Float16* W2T  = (_Float16*)alloc((size_t)Dff * Dm * 2);
  _Float16* qh   = (_Float16*)alloc((size_t)BT * Dm * 2);
  _Float16* kh   = (_Float16*)alloc((size_t)BT * Dm * 2);
  _Float16* vt   = (_Float16*)alloc((size_t)Bb * Dm * Tt * 2);
  float*    attn = (float*)   alloc((size_t)BT * Dm * 4);
  float*    xbuf = (float*)   alloc((size_t)BT * Dm * 4);
  _Float16* xh   = (_Float16*)alloc((size_t)BT * Dm * 2);
  _Float16* hh   = (_Float16*)alloc((size_t)BT * Dff * 2);
  float*    ybuf = (float*)   alloc((size_t)BT * Dm * 4);
  (void)ws_size; (void)n_in; (void)in_sizes; (void)out_size;

  // 1) converts / transposes
  {
    int n = BT * Dm;
    cvt_f32_to_f16<<<(n + 255) / 256, 256, 0, stream>>>(tokens, tokh, n);
    int nw = Dm * Dm;
    transpose_f16<<<(nw + 255) / 256, 256, 0, stream>>>(Wq, WqT, Dm, Dm);
    transpose_f16<<<(nw + 255) / 256, 256, 0, stream>>>(Wk, WkT, Dm, Dm);
    transpose_f16<<<(nw + 255) / 256, 256, 0, stream>>>(Wv, WvT, Dm, Dm);
    int n1 = Dm * Dff;
    transpose_f16<<<(n1 + 255) / 256, 256, 0, stream>>>(W1, W1T, Dm, Dff);
    transpose_f16<<<(n1 + 255) / 256, 256, 0, stream>>>(W2, W2T, Dff, Dm);
  }

  // 2) QKV projections (q,k row-major f16; v transposed [b][d][T])
  {
    dim3 grid(BT / 16, Dm / (16 * 8));
    gemm_wmma<0><<<grid, 256, 0, stream>>>(tokh, WqT, bq, qh, BT, Dm, Dm, Tt);
    gemm_wmma<0><<<grid, 256, 0, stream>>>(tokh, WkT, bk, kh, BT, Dm, Dm, Tt);
    gemm_wmma<1><<<grid, 256, 0, stream>>>(tokh, WvT, bv, vt, BT, Dm, Dm, Tt);
  }

  // 3) flash attention
  attn_flash<<<128, 128, 0, stream>>>(qh, kh, vt, dscale, attn);

  // 4) LN1: x = LN(tokens + attn), also f16 copy for FFN
  ln_rows<<<BT / 8, 256, 0, stream>>>(attn, tokens, g1, be1, xbuf, xh, BT);

  // 5) FFN
  {
    dim3 g1d(BT / 16, Dff / (16 * 8));
    gemm_wmma<2><<<g1d, 256, 0, stream>>>(xh, W1T, b1, hh, BT, Dff, Dm, Tt);
    dim3 g2d(BT / 16, Dm / (16 * 8));
    gemm_wmma<3><<<g2d, 256, 0, stream>>>(hh, W2T, b2, ybuf, BT, Dm, Dff, Tt);
  }

  // 6) LN2: z = LN(y + x) -> output
  ln_rows<<<BT / 8, 256, 0, stream>>>(ybuf, xbuf, g2, be2, (float*)d_out, nullptr, BT);
}
